// SlotEncoder_37778532336202
// MI455X (gfx1250) — matrix operand order repaired and checked by reading, not compiled, not run on hardware
//
#include <hip/hip_runtime.h>
#include <hip/hip_bf16.h>
#include <math.h>

// ---------------- problem constants ----------------
#define NBATCH   32
#define NTOK     4096
#define D_IN     384
#define D_SLOT   128
#define NSLOT    8
#define MLP_HID  256
#define NCHUNK   16          // attention n-chunks per batch
#define CHUNK    256         // tokens per chunk
#define QSCALE   0.08838834764831845f   // 128^-0.5
#define EPS_ATT  1e-8f
#define LN_EPS   1e-5f
#define THRESH   0.9f

// ---------------- CDNA5 WMMA types ----------------
typedef __bf16 v16bf __attribute__((ext_vector_type(16)));
typedef __bf16 v8bf  __attribute__((ext_vector_type(8)));
typedef float  v8f   __attribute__((ext_vector_type(8)));

__device__ __forceinline__ unsigned short f2bf(float x) {
  unsigned int u = __float_as_uint(x);
  u += 0x7FFFu + ((u >> 16) & 1u);           // round-to-nearest-even
  return (unsigned short)(u >> 16);
}
__device__ __forceinline__ float bf2f(unsigned short h) {
  return __uint_as_float(((unsigned int)h) << 16);
}

// A-matrix 16x32 bf16 fragment (ISA 7.12.2): per lane, K = base+{0..7} and
// base+{16..23} (lanes 0-15) or +8 / +24 (lanes 16-31). Caller passes the
// per-lane base pointer; chunks live at +0 and +16 elements (16B aligned).
__device__ __forceinline__ v16bf load_fragA(const unsigned short* p) {
  v8bf lo = *(const v8bf*)(p);
  v8bf hi = *(const v8bf*)(p + 16);
  v16bf r;
#pragma unroll
  for (int i = 0; i < 8; ++i) { r[i] = lo[i]; r[i + 8] = hi[i]; }
  return r;
}
// B-matrix 32x16 bf16 fragment: per lane (column n), K = base+{0..7} and
// base+{16..23} (lanes 0-15) or +8 / +24 (lanes 16-31), same as A.
__device__ __forceinline__ v16bf load_fragB(const unsigned short* p) {
  v8bf lo = *(const v8bf*)(p);
  v8bf hi = *(const v8bf*)(p + 16);
  v16bf r;
#pragma unroll
  for (int i = 0; i < 8; ++i) { r[i] = lo[i]; r[i + 8] = hi[i]; }
  return r;
}

// ---------------- kernel 0: transpose+convert Wk/Wv -> bf16 [128][384] ------
__global__ void prep_weights(const float* __restrict__ Wk, const float* __restrict__ Wv,
                             unsigned short* __restrict__ WkT, unsigned short* __restrict__ WvT) {
  int idx = blockIdx.x * 256 + threadIdx.x;
  if (idx >= 2 * D_IN * D_SLOT) return;
  int mtx = idx / (D_IN * D_SLOT);
  int r   = idx % (D_IN * D_SLOT);
  int n = r / D_IN;            // output row = column of W
  int k = r % D_IN;
  const float* W = mtx ? Wv : Wk;
  unsigned short* O = mtx ? WvT : WkT;
  O[n * D_IN + k] = f2bf(W[k * D_SLOT + n]);
}

// ---------------- kernel 1: fused LN + K/V projection (WMMA bf16) -----------
// grid = 8192 blocks of 16 feature rows; block = 256 threads (8 waves).
// Wave w computes 16x16 output tiles for columns [16w,16w+16) of both K and V.
__global__ __launch_bounds__(256)
void proj_kernel(const float* __restrict__ feat,
                 const float* __restrict__ g, const float* __restrict__ bvec,
                 const unsigned short* __restrict__ WkT,
                 const unsigned short* __restrict__ WvT,
                 unsigned short* __restrict__ kout,
                 unsigned short* __restrict__ vout) {
  __shared__ unsigned short A[16][392];   // 392 stride -> conflict-free b128 LDS reads
  const int t = threadIdx.x;
  const int row = t >> 4, sub = t & 15;
  const long grow = (long)blockIdx.x * 16 + row;

  // ---- LayerNorm over 384, 16 lanes per row, values kept in registers ----
  float vals[24];
  float s0 = 0.f, s1 = 0.f;
#pragma unroll
  for (int i = 0; i < 24; ++i) {
    float x = feat[grow * D_IN + sub + 16 * i];
    vals[i] = x; s0 += x; s1 += x * x;
  }
#pragma unroll
  for (int off = 8; off; off >>= 1) { s0 += __shfl_xor(s0, off, 16); s1 += __shfl_xor(s1, off, 16); }
  float mean = s0 * (1.f / 384.f);
  float rstd = rsqrtf(s1 * (1.f / 384.f) - mean * mean + LN_EPS);
#pragma unroll
  for (int i = 0; i < 24; ++i) {
    int col = sub + 16 * i;
    A[row][col] = f2bf((vals[i] - mean) * rstd * g[col] + bvec[col]);
  }
  __syncthreads();

  // ---- WMMA: acc(16x16) over K=384 (12 steps of 32), for Wk and Wv ----
  const int w = t >> 5, lane = t & 31;
  const int m = lane & 15, half = lane >> 4;
  v8f accK, accV;
#pragma unroll
  for (int i = 0; i < 8; ++i) { accK[i] = 0.f; accV[i] = 0.f; }
  const unsigned short* arow  = &A[m][0];
  const unsigned short* bkrow = WkT + (w * 16 + m) * D_IN;
  const unsigned short* bvrow = WvT + (w * 16 + m) * D_IN;
#pragma unroll
  for (int kk = 0; kk < 12; ++kk) {
    v16bf af = load_fragA(arow + kk * 32 + half * 8);
    v16bf bk = load_fragB(bkrow + kk * 32 + half * 8);
    v16bf bv = load_fragB(bvrow + kk * 32 + half * 8);
    accK = __builtin_amdgcn_wmma_f32_16x16x32_bf16(false, af, false, bk, (short)0, accK, false, false);
    accV = __builtin_amdgcn_wmma_f32_16x16x32_bf16(false, af, false, bv, (short)0, accV, false, false);
  }

  // C layout: VGPR i = row (i | half*8), N = lane&15
  const long row0 = (long)blockIdx.x * 16;
  const int d = w * 16 + m;
#pragma unroll
  for (int i = 0; i < 8; ++i) {
    long orow = row0 + i + (half ? 8 : 0);
    kout[orow * D_SLOT + d] = f2bf(accK[i]);
    vout[orow * D_SLOT + d] = f2bf(accV[i]);
  }
}

// ---------------- kernel 2: slots <- broadcast slot_mu ----------------------
__global__ void init_slots(const float* __restrict__ mu, float* __restrict__ slots) {
  int idx = blockIdx.x * 256 + threadIdx.x;
  if (idx < NBATCH * NSLOT * D_SLOT) slots[idx] = mu[idx & (D_SLOT - 1)];
}

// ---------------- kernel 3: LN(slots) and q = s_ln @ Wq (bf16, padded 16) ---
__global__ __launch_bounds__(128)
void slot_pre(const float* __restrict__ slots,
              const float* __restrict__ g, const float* __restrict__ b,
              const float* __restrict__ Wq, unsigned short* __restrict__ qbuf) {
  __shared__ float lnS[NSLOT][D_SLOT];
  const int bi = blockIdx.x;
  const int t = threadIdx.x;
  const int row = t >> 4, sub = t & 15;
  float vals[8]; float s0 = 0.f, s1 = 0.f;
#pragma unroll
  for (int i = 0; i < 8; ++i) {
    float x = slots[((long)bi * NSLOT + row) * D_SLOT + sub + 16 * i];
    vals[i] = x; s0 += x; s1 += x * x;
  }
#pragma unroll
  for (int off = 8; off; off >>= 1) { s0 += __shfl_xor(s0, off, 16); s1 += __shfl_xor(s1, off, 16); }
  float mean = s0 * (1.f / 128.f);
  float rstd = rsqrtf(s1 * (1.f / 128.f) - mean * mean + LN_EPS);
#pragma unroll
  for (int i = 0; i < 8; ++i) {
    int col = sub + 16 * i;
    lnS[row][col] = (vals[i] - mean) * rstd * g[col] + b[col];
  }
  __syncthreads();
  const int d = t;
#pragma unroll
  for (int s = 0; s < NSLOT; ++s) {
    float acc = 0.f;
    for (int k = 0; k < D_SLOT; ++k) acc += lnS[s][k] * Wq[k * D_SLOT + d];
    qbuf[((long)bi * 16 + s) * D_SLOT + d] = f2bf(acc);
    qbuf[((long)bi * 16 + 8 + s) * D_SLOT + d] = 0;   // pad rows 8..15
  }
}

// ---------------- kernel 4: attention pass (WMMA logits + softmax + partials)
// grid = (chunk=16, batch=32); block = 256 (8 waves, each 2 N-tiles of 16).
__global__ __launch_bounds__(256)
void attn_kernel(const unsigned short* __restrict__ qbuf,
                 const unsigned short* __restrict__ kbuf,
                 const unsigned short* __restrict__ vbuf,
                 float* __restrict__ nump, float* __restrict__ denp,
                 float* __restrict__ attn_out, int write_attn) {
  __shared__ float attnS[NSLOT][CHUNK];
  const int chunk = blockIdx.x, bi = blockIdx.y;
  const int n0 = chunk * CHUNK;
  const int t = threadIdx.x, w = t >> 5, lane = t & 31;
  const int m = lane & 15, half = lane >> 4;

  // A fragments: q[b] 16x128 bf16, reused for both tiles
  const unsigned short* qrow = qbuf + ((long)bi * 16 + m) * D_SLOT;
  v16bf aF[4];
#pragma unroll
  for (int kk = 0; kk < 4; ++kk) aF[kk] = load_fragA(qrow + kk * 32 + half * 8);

#pragma unroll
  for (int tt = 0; tt < 2; ++tt) {
    const int tileN = w * 2 + tt;
    const long nrow = (long)bi * NTOK + n0 + tileN * 16 + m;
    const unsigned short* kb = kbuf + nrow * D_SLOT;   // row n of k: contiguous over d
    v8f acc;
#pragma unroll
    for (int i = 0; i < 8; ++i) acc[i] = 0.f;
#pragma unroll
    for (int kk = 0; kk < 4; ++kk) {
      v16bf bf = load_fragB(kb + kk * 32 + half * 8);
      acc = __builtin_amdgcn_wmma_f32_16x16x32_bf16(false, aF[kk], false, bf, (short)0, acc, false, false);
    }
    // slot s logits live in acc[s] for lanes 0..15 (M=0..7); softmax over s
    if (half == 0) {
      float l[8], mx = -1e30f;
#pragma unroll
      for (int i = 0; i < 8; ++i) { l[i] = acc[i] * QSCALE; mx = fmaxf(mx, l[i]); }
      float sum = 0.f;
#pragma unroll
      for (int i = 0; i < 8; ++i) { l[i] = __expf(l[i] - mx); sum += l[i]; }
      float inv = 1.f / sum;
      const int nl = tileN * 16 + m;
#pragma unroll
      for (int i = 0; i < 8; ++i) {
        float a = l[i] * inv;
        attnS[i][nl] = a;
        if (write_attn) attn_out[((long)bi * NSLOT + i) * NTOK + n0 + nl] = a;
      }
    }
  }
  __syncthreads();

  // den partial: wave w reduces slot s=w over this chunk (deterministic store)
  {
    float s = 0.f;
#pragma unroll
    for (int j = 0; j < 8; ++j) s += attnS[w][lane + 32 * j];
#pragma unroll
    for (int off = 16; off; off >>= 1) s += __shfl_xor(s, off, 32);
    if (lane == 0) denp[((long)bi * NCHUNK + chunk) * NSLOT + w] = s;
  }
  // num partial: thread owns (4 slots, 1 dim), loop over 256 tokens of chunk
  {
    const int d  = t & (D_SLOT - 1);
    const int sb = (t >> 7) * 4;
    float a0 = 0.f, a1 = 0.f, a2 = 0.f, a3 = 0.f;
    for (int n = 0; n < CHUNK; ++n) {
      float vv = bf2f(vbuf[((long)bi * NTOK + n0 + n) * D_SLOT + d]);
      a0 += attnS[sb + 0][n] * vv;
      a1 += attnS[sb + 1][n] * vv;
      a2 += attnS[sb + 2][n] * vv;
      a3 += attnS[sb + 3][n] * vv;
    }
    float* np = nump + ((long)bi * NCHUNK + chunk) * NSLOT * D_SLOT;
    np[(sb + 0) * D_SLOT + d] = a0;
    np[(sb + 1) * D_SLOT + d] = a1;
    np[(sb + 2) * D_SLOT + d] = a2;
    np[(sb + 3) * D_SLOT + d] = a3;
  }
}

// ---------------- kernel 5: GRU cell + LN + MLP residual per (batch,slot) ---
__global__ __launch_bounds__(128)
void slot_update(const float* __restrict__ nump, const float* __restrict__ denp,
                 float* __restrict__ slots,
                 const float* __restrict__ W_ih, const float* __restrict__ W_hh,
                 const float* __restrict__ b_ih, const float* __restrict__ b_hh,
                 const float* __restrict__ gm, const float* __restrict__ bm,
                 const float* __restrict__ W1, const float* __restrict__ b1,
                 const float* __restrict__ W2, const float* __restrict__ b2,
                 float* __restrict__ raw_out, int write_raw) {
  __shared__ float xs[D_SLOT], hs[D_SLOT], hl[D_SLOT], hidv[MLP_HID];
  __shared__ float r0[4], r1[4];
  const int bi = blockIdx.x >> 3, s = blockIdx.x & 7;
  const int d = threadIdx.x;

  // combine chunk partials: updates = num / (den + eps)
  float den = 0.f, nm = 0.f;
#pragma unroll
  for (int c = 0; c < NCHUNK; ++c) {
    den += denp[((long)bi * NCHUNK + c) * NSLOT + s];
    nm  += nump[(((long)bi * NCHUNK + c) * NSLOT + s) * D_SLOT + d];
  }
  float x = nm / (den + EPS_ATT);
  float h = slots[((long)bi * NSLOT + s) * D_SLOT + d];
  xs[d] = x; hs[d] = h;
  __syncthreads();

  // GRU gates (torch order r,z,n); gx = x @ W_ih^T + b_ih, gh = h @ W_hh^T + b_hh
  float gxr = b_ih[d], gxz = b_ih[D_SLOT + d], gxn = b_ih[2 * D_SLOT + d];
  float ghr = b_hh[d], ghz = b_hh[D_SLOT + d], ghn = b_hh[2 * D_SLOT + d];
  for (int k = 0; k < D_SLOT; ++k) {
    float xk = xs[k], hk = hs[k];
    gxr += xk * W_ih[d * D_SLOT + k];
    gxz += xk * W_ih[(D_SLOT + d) * D_SLOT + k];
    gxn += xk * W_ih[(2 * D_SLOT + d) * D_SLOT + k];
    ghr += hk * W_hh[d * D_SLOT + k];
    ghz += hk * W_hh[(D_SLOT + d) * D_SLOT + k];
    ghn += hk * W_hh[(2 * D_SLOT + d) * D_SLOT + k];
  }
  float r  = 1.f / (1.f + __expf(-(gxr + ghr)));
  float z  = 1.f / (1.f + __expf(-(gxz + ghz)));
  float nn = tanhf(gxn + r * ghn);
  float newh = (1.f - z) * nn + z * h;

  // LayerNorm(newh) over 128 (4-wave block reduce)
  float a0 = newh, a1 = newh * newh;
#pragma unroll
  for (int off = 16; off; off >>= 1) { a0 += __shfl_xor(a0, off, 32); a1 += __shfl_xor(a1, off, 32); }
  if ((d & 31) == 0) { r0[d >> 5] = a0; r1[d >> 5] = a1; }
  __syncthreads();
  float S0 = r0[0] + r0[1] + r0[2] + r0[3];
  float S1 = r1[0] + r1[1] + r1[2] + r1[3];
  float mean = S0 * (1.f / 128.f);
  float rstd = rsqrtf(S1 * (1.f / 128.f) - mean * mean + LN_EPS);
  hl[d] = (newh - mean) * rstd * gm[d] + bm[d];
  __syncthreads();

  // MLP: silu(h_ln @ W1 + b1) @ W2 + b2, residual
  float h0 = b1[d], h1 = b1[D_SLOT + d];
  for (int k = 0; k < D_SLOT; ++k) {
    float hk = hl[k];
    h0 += hk * W1[k * MLP_HID + d];
    h1 += hk * W1[k * MLP_HID + D_SLOT + d];
  }
  h0 = h0 / (1.f + __expf(-h0));
  h1 = h1 / (1.f + __expf(-h1));
  hidv[d] = h0; hidv[D_SLOT + d] = h1;
  __syncthreads();
  float o = b2[d];
  for (int j = 0; j < MLP_HID; ++j) o += hidv[j] * W2[j * D_SLOT + d];
  float outv = newh + o;
  slots[((long)bi * NSLOT + s) * D_SLOT + d] = outv;
  if (write_raw) raw_out[((long)bi * NSLOT + s) * D_SLOT + d] = outv;
}

// ---------------- kernel 6: greedy merge map + segment-mean merge -----------
__global__ __launch_bounds__(128)
void merge_kernel(const float* __restrict__ slots,
                  float* __restrict__ merged, float* __restrict__ mmap) {
  __shared__ float sl[NSLOT][D_SLOT];
  __shared__ float simM[64];
  __shared__ float nrm[NSLOT];
  __shared__ int   mt[NSLOT];
  const int bi = blockIdx.x;
  const int t = threadIdx.x;
  for (int i = t; i < NSLOT * D_SLOT; i += 128)
    sl[i >> 7][i & 127] = slots[(long)bi * NSLOT * D_SLOT + i];
  __syncthreads();
  {
    const int row = t >> 4, sub = t & 15;
    float ss = 0.f;
#pragma unroll
    for (int i = 0; i < 8; ++i) { float v = sl[row][sub + 16 * i]; ss += v * v; }
#pragma unroll
    for (int off = 8; off; off >>= 1) ss += __shfl_xor(ss, off, 16);
    if (sub == 0) nrm[row] = sqrtf(ss);
  }
  __syncthreads();
  if (t < 64) {
    int i = t >> 3, j = t & 7;
    float dot = 0.f;
    for (int k = 0; k < D_SLOT; ++k) dot += sl[i][k] * sl[j][k];
    float sim = dot / (fmaxf(nrm[i], 1e-12f) * fmaxf(nrm[j], 1e-12f));
    if (i == j) sim -= 2.f;
    simM[t] = sim;
  }
  if (t < NSLOT) mt[t] = t;
  __syncthreads();
  if (t == 0) {
    for (int it = 0; it < NSLOT; ++it) {
      float mx = -1e30f; int idx = 0;
      for (int q = 0; q < 64; ++q) if (simM[q] > mx) { mx = simM[q]; idx = q; } // first max
      if (mx > THRESH) {
        int rr = idx >> 3, cc = idx & 7;
        int src = rr > cc ? rr : cc, tgt = rr > cc ? cc : rr;
        mt[src] = tgt;
        for (int q = 0; q < 64; ++q) if ((q >> 3) == src || (q & 7) == src) simM[q] = -2.f;
      }
    }
  }
  __syncthreads();
#pragma unroll
  for (int j = 0; j < NSLOT; ++j) {
    float sum = 0.f; int c = 0;
#pragma unroll
    for (int i = 0; i < NSLOT; ++i) if (mt[i] == j) { sum += sl[i][t]; ++c; }
    merged[((long)bi * NSLOT + j) * D_SLOT + t] = sum / (float)(c > 0 ? c : 1);
  }
  if (t < NSLOT) mmap[bi * NSLOT + t] = (float)mt[t];
}

// ---------------- host side -------------------------------------------------
static const size_t OFF_WKT = 0;
static const size_t OFF_WVT = OFF_WKT + (size_t)D_IN * D_SLOT * 2;
static const size_t OFF_K   = OFF_WVT + (size_t)D_IN * D_SLOT * 2;
static const size_t OFF_V   = OFF_K + (size_t)NBATCH * NTOK * D_SLOT * 2;
static const size_t OFF_Q   = OFF_V + (size_t)NBATCH * NTOK * D_SLOT * 2;
static const size_t OFF_SL  = OFF_Q + (size_t)NBATCH * 16 * D_SLOT * 2;
static const size_t OFF_NP  = OFF_SL + (size_t)NBATCH * NSLOT * D_SLOT * 4;
static const size_t OFF_DP  = OFF_NP + (size_t)NBATCH * NCHUNK * NSLOT * D_SLOT * 4;

// d_out layout (floats): merged | attn | merge_map | raw_slots
static const size_t OUT_MERGED = 0;
static const size_t OUT_ATTN   = (size_t)NBATCH * NSLOT * D_SLOT;                 // 32768
static const size_t OUT_MAP    = OUT_ATTN + (size_t)NBATCH * NSLOT * NTOK;        // 1081344
static const size_t OUT_RAW    = OUT_MAP + (size_t)NBATCH * NSLOT;                // 1081600

extern "C" void kernel_launch(void* const* d_in, const int* in_sizes, int n_in,
                              void* d_out, int out_size, void* d_ws, size_t ws_size,
                              hipStream_t stream) {
  (void)in_sizes; (void)n_in; (void)out_size; (void)ws_size;
  const float* features = (const float*)d_in[0];
  const float* ln_in_g  = (const float*)d_in[1];
  const float* ln_in_b  = (const float*)d_in[2];
  const float* Wk       = (const float*)d_in[3];
  const float* Wv       = (const float*)d_in[4];
  const float* Wq       = (const float*)d_in[5];
  const float* ln_s_g   = (const float*)d_in[6];
  const float* ln_s_b   = (const float*)d_in[7];
  const float* W_ih     = (const float*)d_in[8];
  const float* W_hh     = (const float*)d_in[9];
  const float* b_ih     = (const float*)d_in[10];
  const float* b_hh     = (const float*)d_in[11];
  const float* ln_m_g   = (const float*)d_in[12];
  const float* ln_m_b   = (const float*)d_in[13];
  const float* W1       = (const float*)d_in[14];
  const float* b1       = (const float*)d_in[15];
  const float* W2       = (const float*)d_in[16];
  const float* b2       = (const float*)d_in[17];
  const float* slot_mu  = (const float*)d_in[18];

  char* ws = (char*)d_ws;
  unsigned short* WkT  = (unsigned short*)(ws + OFF_WKT);
  unsigned short* WvT  = (unsigned short*)(ws + OFF_WVT);
  unsigned short* kbuf = (unsigned short*)(ws + OFF_K);
  unsigned short* vbuf = (unsigned short*)(ws + OFF_V);
  unsigned short* qbuf = (unsigned short*)(ws + OFF_Q);
  float* slots = (float*)(ws + OFF_SL);
  float* nump  = (float*)(ws + OFF_NP);
  float* denp  = (float*)(ws + OFF_DP);
  float* out = (float*)d_out;

  prep_weights<<<(2 * D_IN * D_SLOT + 255) / 256, 256, 0, stream>>>(Wk, Wv, WkT, WvT);
  proj_kernel<<<(NBATCH * NTOK) / 16, 256, 0, stream>>>(features, ln_in_g, ln_in_b,
                                                        WkT, WvT, kbuf, vbuf);
  init_slots<<<(NBATCH * NSLOT * D_SLOT + 255) / 256, 256, 0, stream>>>(slot_mu, slots);

  for (int it = 0; it < 3; ++it) {
    const int last = (it == 2) ? 1 : 0;
    slot_pre<<<NBATCH, 128, 0, stream>>>(slots, ln_s_g, ln_s_b, Wq, qbuf);
    attn_kernel<<<dim3(NCHUNK, NBATCH), 256, 0, stream>>>(qbuf, kbuf, vbuf, nump, denp,
                                                          out + OUT_ATTN, last);
    slot_update<<<NBATCH * NSLOT, 128, 0, stream>>>(nump, denp, slots,
                                                    W_ih, W_hh, b_ih, b_hh,
                                                    ln_m_g, ln_m_b, W1, b1, W2, b2,
                                                    out + OUT_RAW, last);
  }
  merge_kernel<<<NBATCH, 128, 0, stream>>>(slots, out + OUT_MERGED, out + OUT_MAP);
}
